// VectorQuantizer_44727789421010
// MI455X (gfx1250) — matrix operand order repaired and checked by reading, not compiled, not run on hardware
//
#include <hip/hip_runtime.h>
#include <stdint.h>

#define Kc     4096
#define Dc     256
#define NTOK   16384
#define MT     64            // tokens per workgroup
#define NTILE  64            // codes per LDS chunk (double-buffered)
#define NCHUNK (Kc / NTILE)  // 64
#define LROW   264           // padded LDS row (bf16 elems): 256 + 8 (16B pad)
#define LROWB  528           // row bytes

typedef __bf16        v16bf __attribute__((ext_vector_type(16)));
typedef float         v8f   __attribute__((ext_vector_type(8)));
typedef unsigned int  u32x4 __attribute__((ext_vector_type(4)));

union FragBF { v16bf v; u32x4 q[2]; };

__device__ __forceinline__ unsigned short f2bf_rne(float f) {
    unsigned u = __float_as_uint(f);
    unsigned r = u + 0x7FFFu + ((u >> 16) & 1u);
    return (unsigned short)(r >> 16);
}

// raw 32-bit LDS offset of a __shared__ object (generic -> addrspace(3) -> int)
__device__ __forceinline__ unsigned ldsoff(const void* p) {
    return (unsigned)(unsigned long long)(const __attribute__((address_space(3))) char*)p;
}

// ---------- prep: codebook -> bf16 hi/lo planes + ||c||^2 ----------
__global__ void __launch_bounds__(256) vq_prep(const float* __restrict__ cb,
                                               unsigned short* __restrict__ hi,
                                               unsigned short* __restrict__ lo,
                                               float* __restrict__ cnorm) {
    __shared__ float red[256];
    const int k = blockIdx.x, d = threadIdx.x;
    float f = cb[k * Dc + d];
    unsigned short h = f2bf_rne(f);
    float fh = __uint_as_float(((unsigned)h) << 16);
    unsigned short l = f2bf_rne(f - fh);
    hi[k * Dc + d] = h;
    lo[k * Dc + d] = l;
    red[d] = f * f;
    __syncthreads();
    for (int s = 128; s > 0; s >>= 1) {
        if (d < s) red[d] += red[d + s];
        __syncthreads();
    }
    if (d == 0) cnorm[k] = red[0];
}

// ---------- main: distance GEMM (bf16 split, f32 acc) + argmin + gather ----------
__global__ void __launch_bounds__(256) vq_main(const float* __restrict__ x,
                                               const float* __restrict__ cb,
                                               const unsigned short* __restrict__ cbhi,
                                               const unsigned short* __restrict__ cblo,
                                               const float* __restrict__ cnorm,
                                               float* __restrict__ out) {
    __shared__ __align__(16) unsigned short sAhi[MT * LROW];
    __shared__ __align__(16) unsigned short sAlo[MT * LROW];
    __shared__ __align__(16) unsigned short sBhi[2][NTILE * LROW];
    __shared__ __align__(16) unsigned short sBlo[2][NTILE * LROW];
    __shared__ unsigned long long best[MT];
    __shared__ int sidx[MT];

    const int tid = threadIdx.x;
    const int t0  = blockIdx.x * MT;  // first token of this tile
    const int b   = t0 >> 10;         // 1024 tokens per batch image
    const int hw0 = t0 & 1023;
    const float* xb = x + (size_t)b * Dc * 1024 + hw0;

    if (tid < MT) best[tid] = ~0ull;

    // issue async fill of chunk 0 into buffer 0 (overlaps the A-tile conversion below)
    // each chunk/plane: 64 rows * 512B = 2048 b128 -> 8 per thread per plane
    auto issue_chunk = [&](int chunk, int buf) {
        unsigned baseH = ldsoff(&sBhi[buf][0]);
        unsigned baseL = ldsoff(&sBlo[buf][0]);
        unsigned g0 = (unsigned)chunk * (unsigned)(NTILE * Dc * 2);  // bytes
#pragma unroll
        for (int i = 0; i < 8; ++i) {
            int v = i * 256 + tid;
            int row = v >> 5, col = v & 31;
            unsigned lds = (unsigned)(row * LROWB + col * 16);
            unsigned g   = g0 + (unsigned)v * 16u;
            asm volatile("global_load_async_to_lds_b128 %0, %1, %2"
                         :: "v"(baseH + lds), "v"(g), "s"(cbhi) : "memory");
            asm volatile("global_load_async_to_lds_b128 %0, %1, %2"
                         :: "v"(baseL + lds), "v"(g), "s"(cblo) : "memory");
        }
    };
    issue_chunk(0, 0);

    // A tile: load [64 tok x 256 feat] (feature stride 1024 in x), split to bf16 hi/lo
    for (int j = 0; j < 64; ++j) {
        int d   = (j << 2) + (tid >> 6);
        int tok = tid & 63;
        float f = xb[d * 1024 + tok];
        unsigned short h = f2bf_rne(f);
        float fh = __uint_as_float(((unsigned)h) << 16);
        unsigned short l = f2bf_rne(f - fh);
        sAhi[tok * LROW + d] = h;
        sAlo[tok * LROW + d] = l;
    }

    const int lane    = tid & 31;
    const int laneM   = lane & 15;
    const int hf      = lane >> 4;
    const int wid     = tid >> 5;
    const int mBase   = (wid & 3) << 4;   // 4 M-blocks of 16
    const int colBase = (wid >> 2) << 5;  // 2 N-columns of 32 (2 tiles each)

    float runVal[8];
    int   runIdx[8];
#pragma unroll
    for (int r = 0; r < 8; ++r) { runVal[r] = 3.4e38f; runIdx[r] = 0; }

    for (int chunk = 0; chunk < NCHUNK; ++chunk) {
        const int cur = chunk & 1;
        // wait for this chunk's async fill, make it visible to all waves
        asm volatile("s_wait_asynccnt 0x0" ::: "memory");
        __syncthreads();
        // kick off next chunk's fill into the other buffer (overlaps compute)
        if (chunk + 1 < NCHUNK) issue_chunk(chunk + 1, (chunk + 1) & 1);

        float cn[2];
#pragma unroll
        for (int t = 0; t < 2; ++t)
            cn[t] = cnorm[chunk * NTILE + colBase + t * 16 + laneM];

        v8f acc[2];
#pragma unroll
        for (int t = 0; t < 2; ++t)
#pragma unroll
            for (int r = 0; r < 8; ++r) acc[t][r] = 0.0f;

        const unsigned short* bufH = &sBhi[cur][0];
        const unsigned short* bufL = &sBlo[cur][0];
#pragma unroll
        for (int ks = 0; ks < Dc; ks += 32) {
            FragBF ahi, alo;
            {
                const unsigned short* bah = &sAhi[(mBase + laneM) * LROW + ks + hf * 8];
                ahi.q[0] = *(const u32x4*)bah;
                ahi.q[1] = *(const u32x4*)(bah + 16);
                const unsigned short* bal = &sAlo[(mBase + laneM) * LROW + ks + hf * 8];
                alo.q[0] = *(const u32x4*)bal;
                alo.q[1] = *(const u32x4*)(bal + 16);
            }
#pragma unroll
            for (int t = 0; t < 2; ++t) {
                FragBF bhi, blo;
                const unsigned short* bbh = &bufH[(colBase + t * 16 + laneM) * LROW + ks + hf * 8];
                bhi.q[0] = *(const u32x4*)bbh;
                bhi.q[1] = *(const u32x4*)(bbh + 16);
                const unsigned short* bbl = &bufL[(colBase + t * 16 + laneM) * LROW + ks + hf * 8];
                blo.q[0] = *(const u32x4*)bbl;
                blo.q[1] = *(const u32x4*)(bbl + 16);
                acc[t] = __builtin_amdgcn_wmma_f32_16x16x32_bf16(false, ahi.v, false, bhi.v,
                                                                 (short)0, acc[t], false, false);
                acc[t] = __builtin_amdgcn_wmma_f32_16x16x32_bf16(false, ahi.v, false, blo.v,
                                                                 (short)0, acc[t], false, false);
                acc[t] = __builtin_amdgcn_wmma_f32_16x16x32_bf16(false, alo.v, false, bhi.v,
                                                                 (short)0, acc[t], false, false);
            }
        }
        // fold chunk distances into per-lane running argmin (strict < keeps first index)
#pragma unroll
        for (int r = 0; r < 8; ++r) {
#pragma unroll
            for (int t = 0; t < 2; ++t) {
                float val = cn[t] - 2.0f * acc[t][r];
                int   n   = chunk * NTILE + colBase + t * 16 + laneM;
                if (val < runVal[r]) { runVal[r] = val; runIdx[r] = n; }
            }
        }
    }

    // merge: orderable-float key (high) | index (low) via 64-bit LDS atomicMin
#pragma unroll
    for (int r = 0; r < 8; ++r) {
        int m = mBase + hf * 8 + r;  // C layout: VGPR r -> M = 8*(lane/16)+r
        unsigned u = __float_as_uint(runVal[r]);
        u = (u & 0x80000000u) ? ~u : (u | 0x80000000u);
        unsigned long long packed = (((unsigned long long)u) << 32) | (unsigned)runIdx[r];
        atomicMin(&best[m], packed);
    }
    __syncthreads();
    if (tid < MT) sidx[tid] = (int)(best[tid] & 0xFFFFFFFFull);
    __syncthreads();

    // gather codebook rows (f32) into [B, D, H, W] output, coalesced on hw
    float* ob = out + (size_t)b * Dc * 1024 + hw0;
    for (int j = 0; j < 64; ++j) {
        int d   = (j << 2) + (tid >> 6);
        int tok = tid & 63;
        ob[d * 1024 + tok] = cb[(size_t)sidx[tok] * Dc + d];
    }
}

extern "C" void kernel_launch(void* const* d_in, const int* in_sizes, int n_in,
                              void* d_out, int out_size, void* d_ws, size_t ws_size,
                              hipStream_t stream) {
    const float* x  = (const float*)d_in[0];
    const float* cb = (const float*)d_in[1];

    unsigned short* cbhi  = (unsigned short*)d_ws;             // 2 MB
    unsigned short* cblo  = cbhi + (size_t)Kc * Dc;            // 2 MB
    float*          cnorm = (float*)(cblo + (size_t)Kc * Dc);  // 16 KB
    float*          out   = (float*)d_out;

    vq_prep<<<Kc, 256, 0, stream>>>(cb, cbhi, cblo, cnorm);
    vq_main<<<NTOK / MT, 256, 0, stream>>>(x, cb, cbhi, cblo, cnorm, out);
}
